// GIN_82652350644363
// MI455X (gfx1250) — matrix operand order repaired and checked by reading, not compiled, not run on hardware
//
#include <hip/hip_runtime.h>
#include <cstdint>

#define NN 100000
#define EO 600000
#define EL 1200000

typedef __attribute__((ext_vector_type(16))) _Float16 v16h;
typedef __attribute__((ext_vector_type(8)))  _Float16 v8h;
typedef __attribute__((ext_vector_type(8)))  float    v8f;
typedef __attribute__((ext_vector_type(4)))  unsigned int u32x4;
typedef __attribute__((ext_vector_type(8)))  int i32x8;
typedef __attribute__((ext_vector_type(4)))  int i32x4;

#if defined(__AMDGCN__) && __has_builtin(__builtin_amdgcn_tensor_load_to_lds) && __has_builtin(__builtin_amdgcn_s_wait_tensorcnt)
#define USE_TDM 1
#else
#define USE_TDM 0
#endif

// ---------------------------------------------------------------------------
// TDM: DMA a contiguous block of n_halfs f16 from global into LDS.
// D# group0: count=1, lds_addr, 57-bit global addr, type=2.
// D# group1: data_size=2B, tensor_dim0=n_halfs, tensor/tile dim1=1,
//            tile_dim0=n_halfs (<=65535), dim0 stride=n_halfs.
// ---------------------------------------------------------------------------
#if USE_TDM
__device__ __forceinline__ void tdm_load_lds(const void* gsrc, void* lds_dst, int n_halfs) {
  unsigned long long ga = (unsigned long long)(uintptr_t)gsrc;
  unsigned int la = (unsigned int)(uintptr_t)lds_dst;  // low 32 bits of flat = LDS byte addr
  u32x4 d0;
  d0[0] = 1u;                                              // count=1, user mode
  d0[1] = la;                                              // lds_addr
  d0[2] = (unsigned int)(ga & 0xFFFFFFFFull);              // global_addr[31:0]
  d0[3] = (unsigned int)((ga >> 32) & 0x1FFFFFFull) | (2u << 30); // [56:32] | type=2
  i32x8 d1;
  d1[0] = 0x00010000;                                      // data_size = 2 bytes
  d1[1] = (int)((unsigned)(n_halfs & 0xFFFF) << 16);       // tensor_dim0 lo16
  d1[2] = (int)(((unsigned)n_halfs >> 16) & 0xFFFFu) | (1 << 16); // dim0 hi16 | tensor_dim1=1
  d1[3] = (int)((unsigned)(n_halfs & 0xFFFF) << 16);       // tile_dim0
  d1[4] = 1;                                               // tile_dim1 = 1
  d1[5] = n_halfs;                                         // tensor_dim0_stride lo32
  d1[6] = 0;
  d1[7] = 0;
  i32x4 z4 = {0, 0, 0, 0};
#if __clang_major__ >= 23
  i32x8 z8 = {0, 0, 0, 0, 0, 0, 0, 0};
  __builtin_amdgcn_tensor_load_to_lds(d0, d1, z4, z4, z8, 0);
#else
  __builtin_amdgcn_tensor_load_to_lds(d0, d1, z4, z4, 0);
#endif
}
#endif

__device__ __forceinline__ void stage_weights(const _Float16* __restrict__ g,
                                              _Float16* s, int n_halfs, int tid) {
#if USE_TDM
  if ((tid >> 5) == 0) tdm_load_lds(g, s, n_halfs);   // one TDM issue (wave 0)
#else
  const uint4* gs = (const uint4*)g;
  uint4* ds = (uint4*)s;
  const int n16 = n_halfs >> 3;
  for (int i = tid; i < n16; i += 256) ds[i] = gs[i];
#endif
}

__device__ __forceinline__ v8f zero8() {
  v8f z = {0.f, 0.f, 0.f, 0.f, 0.f, 0.f, 0.f, 0.f};
  return z;
}

// Pack 16 f32 (K = k0..k0+7 and k0+16..k0+23) into a WMMA A fragment slice.
__device__ __forceinline__ void load_a_frag(v16h& a, const float* __restrict__ arow, int k0) {
  float4 f0 = *(const float4*)(arow + k0);
  float4 f1 = *(const float4*)(arow + k0 + 4);
  float4 f2 = *(const float4*)(arow + k0 + 16);
  float4 f3 = *(const float4*)(arow + k0 + 20);
  a[0]=(_Float16)f0.x;  a[1]=(_Float16)f0.y;  a[2]=(_Float16)f0.z;  a[3]=(_Float16)f0.w;
  a[4]=(_Float16)f1.x;  a[5]=(_Float16)f1.y;  a[6]=(_Float16)f1.z;  a[7]=(_Float16)f1.w;
  a[8]=(_Float16)f2.x;  a[9]=(_Float16)f2.y;  a[10]=(_Float16)f2.z; a[11]=(_Float16)f2.w;
  a[12]=(_Float16)f3.x; a[13]=(_Float16)f3.y; a[14]=(_Float16)f3.z; a[15]=(_Float16)f3.w;
}

// ---------------------------------------------------------------------------
// Fused GIN/MLP block:  out = act2( act1(in @ W1 * scale + shift) @ W2 + b2 )
//   act1 = relu (BN folded into scale/shift, includes b1)
//   act2 = relu if final_relu else identity
// 256 threads = 8 waves; each wave computes a 16x128 tile with
// v_wmma_f32_16x16x32_f16 (4 K-steps x 8 N-tiles per GEMM).
// W1t/W2t are f16, pre-transposed [N][K]; staged in LDS via TDM.
// ---------------------------------------------------------------------------
__global__ __launch_bounds__(256, 1)
void fused_mlp_kernel(const float* __restrict__ in, float* __restrict__ out,
                      const _Float16* __restrict__ w1t, const _Float16* __restrict__ w2t,
                      const float* __restrict__ scale, const float* __restrict__ shift,
                      const float* __restrict__ b2, int M, int final_relu)
{
  __shared__ __align__(32) _Float16 sW1[128 * 128];
  __shared__ __align__(32) _Float16 sW2[128 * 128];
  __shared__ __align__(32) _Float16 sH[8][16 * 128];   // per-wave GEMM1->GEMM2 bounce
  __shared__ float sScale[128], sShift[128], sB2[128];

  const int tid  = threadIdx.x;
  const int wave = tid >> 5;
  const int lane = tid & 31;
  const int m    = lane & 15;   // A row within tile / D col
  const int hh   = lane >> 4;   // half-select
  const int nl   = lane & 15;

  stage_weights(w1t, sW1, 128 * 128, tid);
  stage_weights(w2t, sW2, 128 * 128, tid);
  if (tid < 128) { sScale[tid] = scale[tid]; sShift[tid] = shift[tid]; sB2[tid] = b2[tid]; }
#if USE_TDM
  if (wave == 0) __builtin_amdgcn_s_wait_tensorcnt(0);
#endif
  __syncthreads();

  const long rbase = (long)blockIdx.x * 128 + wave * 16;
  long row = rbase + m;
  if (row > (long)M - 1) row = (long)M - 1;   // clamp (stores guarded); keeps EXEC all-ones
  const float* arow = in + row * 128;

  // ---- GEMM1: A from global f32, B = sW1 ----
  v16h a1[4];
#pragma unroll
  for (int ks = 0; ks < 4; ++ks) load_a_frag(a1[ks], arow, ks * 32 + hh * 8);

  v8f acc[8];
#pragma unroll
  for (int nt = 0; nt < 8; ++nt) acc[nt] = zero8();

#pragma unroll
  for (int ks = 0; ks < 4; ++ks) {
#pragma unroll
    for (int nt = 0; nt < 8; ++nt) {
      const v16h b = __builtin_shufflevector(*(const v8h*)(sW1 + ((nt * 16 + nl) * 128 + ks * 32 + hh * 8)), *(const v8h*)(sW1 + ((nt * 16 + nl) * 128 + ks * 32 + hh * 8 + 16)), 0, 1, 2, 3, 4, 5, 6, 7, 8, 9, 10, 11, 12, 13, 14, 15);
      acc[nt] = __builtin_amdgcn_wmma_f32_16x16x32_f16(false, a1[ks], false, b,
                                                       (short)0, acc[nt], false, false);
    }
  }

  // ---- epilogue1: BN(scale/shift) + relu, f16 into per-wave LDS tile ----
  _Float16* hbuf = &sH[wave][0];
#pragma unroll
  for (int nt = 0; nt < 8; ++nt) {
    const int c = nt * 16 + nl;
    const float sc = sScale[c], sf = sShift[c];
#pragma unroll
    for (int r = 0; r < 8; ++r) {
      float t = fmaxf(acc[nt][r] * sc + sf, 0.f);
      hbuf[(r + hh * 8) * 128 + c] = (_Float16)t;
    }
  }
  __syncthreads();

  // ---- GEMM2: A from LDS tile (already f16, row-major), B = sW2 ----
  v16h a2[4];
#pragma unroll
  for (int ks = 0; ks < 4; ++ks) {
    const v8h lo = *(const v8h*)(hbuf + m * 128 + ks * 32 + hh * 8);
    const v8h hi = *(const v8h*)(hbuf + m * 128 + ks * 32 + hh * 8 + 16);
#pragma unroll
    for (int i = 0; i < 8; ++i) { a2[ks][i] = lo[i]; a2[ks][8 + i] = hi[i]; }
  }

  v8f acc2[8];
#pragma unroll
  for (int nt = 0; nt < 8; ++nt) acc2[nt] = zero8();

#pragma unroll
  for (int ks = 0; ks < 4; ++ks) {
#pragma unroll
    for (int nt = 0; nt < 8; ++nt) {
      const v16h b = __builtin_shufflevector(*(const v8h*)(sW2 + ((nt * 16 + nl) * 128 + ks * 32 + hh * 8)), *(const v8h*)(sW2 + ((nt * 16 + nl) * 128 + ks * 32 + hh * 8 + 16)), 0, 1, 2, 3, 4, 5, 6, 7, 8, 9, 10, 11, 12, 13, 14, 15);
      acc2[nt] = __builtin_amdgcn_wmma_f32_16x16x32_f16(false, a2[ks], false, b,
                                                        (short)0, acc2[nt], false, false);
    }
  }

  // ---- epilogue2: +b2 (+relu), store f32 ----
#pragma unroll
  for (int nt = 0; nt < 8; ++nt) {
    const int c = nt * 16 + nl;
    const float bb = sB2[c];
#pragma unroll
    for (int r = 0; r < 8; ++r) {
      const long rr = rbase + r + hh * 8;
      if (rr < M) {
        float t = acc2[nt][r] + bb;
        if (final_relu) t = fmaxf(t, 0.f);
        out[rr * 128 + c] = t;
      }
    }
  }
}

// ---------------------------------------------------------------------------
// Output projection: out[M,64] = in[M,128] @ W + b    (W pre-transposed [64][128] f16)
// ---------------------------------------------------------------------------
__global__ __launch_bounds__(256, 1)
void out_gemm_kernel(const float* __restrict__ in, float* __restrict__ out,
                     const _Float16* __restrict__ wt, const float* __restrict__ ob, int M)
{
  __shared__ __align__(32) _Float16 sW[64 * 128];
  __shared__ float sB[64];

  const int tid  = threadIdx.x;
  const int wave = tid >> 5;
  const int lane = tid & 31;
  const int m    = lane & 15;
  const int hh   = lane >> 4;
  const int nl   = lane & 15;

  stage_weights(wt, sW, 64 * 128, tid);
  if (tid < 64) sB[tid] = ob[tid];
#if USE_TDM
  if (wave == 0) __builtin_amdgcn_s_wait_tensorcnt(0);
#endif
  __syncthreads();

  const long rbase = (long)blockIdx.x * 128 + wave * 16;
  long row = rbase + m;
  if (row > (long)M - 1) row = (long)M - 1;
  const float* arow = in + row * 128;

  v16h a[4];
#pragma unroll
  for (int ks = 0; ks < 4; ++ks) load_a_frag(a[ks], arow, ks * 32 + hh * 8);

  v8f acc[4];
#pragma unroll
  for (int nt = 0; nt < 4; ++nt) acc[nt] = zero8();

#pragma unroll
  for (int ks = 0; ks < 4; ++ks) {
#pragma unroll
    for (int nt = 0; nt < 4; ++nt) {
      const v16h b = __builtin_shufflevector(*(const v8h*)(sW + ((nt * 16 + nl) * 128 + ks * 32 + hh * 8)), *(const v8h*)(sW + ((nt * 16 + nl) * 128 + ks * 32 + hh * 8 + 16)), 0, 1, 2, 3, 4, 5, 6, 7, 8, 9, 10, 11, 12, 13, 14, 15);
      acc[nt] = __builtin_amdgcn_wmma_f32_16x16x32_f16(false, a[ks], false, b,
                                                       (short)0, acc[nt], false, false);
    }
  }

#pragma unroll
  for (int nt = 0; nt < 4; ++nt) {
    const int c = nt * 16 + nl;
    const float bb = sB[c];
#pragma unroll
    for (int r = 0; r < 8; ++r) {
      const long rr = rbase + r + hh * 8;
      if (rr < M) out[rr * 64 + c] = acc[nt][r] + bb;
    }
  }
}

// ---------------------------------------------------------------------------
// Utility kernels
// ---------------------------------------------------------------------------
__global__ void copy_vec4_kernel(float4* __restrict__ dst, const float4* __restrict__ src, long n4) {
  long i = (long)blockIdx.x * blockDim.x + threadIdx.x;
  if (i < n4) dst[i] = src[i];
}

// sum[dst[e]] += x[src[e]]   (sum pre-initialized with x itself -> x + aggregate)
__global__ void scatter_add_kernel(float* __restrict__ sum, const float* __restrict__ x,
                                   const int* __restrict__ src, const int* __restrict__ dst, int E) {
  long t = (long)blockIdx.x * blockDim.x + threadIdx.x;
  int e = (int)(t >> 5);
  if (e >= E) return;
  int c = ((int)t & 31) * 4;
  const float4 v = *(const float4*)(x + (long)src[e] * 128 + c);
  float* d = sum + (long)dst[e] * 128 + c;
  unsafeAtomicAdd(d + 0, v.x);
  unsafeAtomicAdd(d + 1, v.y);
  unsafeAtomicAdd(d + 2, v.z);
  unsafeAtomicAdd(d + 3, v.w);
}

// h[e] = y[p0[e]] + y[p1[e]]
__global__ void pair_gather_kernel(const float* __restrict__ y, const int* __restrict__ pairs,
                                   float* __restrict__ h, int E) {
  long t = (long)blockIdx.x * blockDim.x + threadIdx.x;
  int e = (int)(t >> 5);
  if (e >= E) return;
  int c = ((int)t & 31) * 4;
  const long p0 = pairs[2 * e], p1 = pairs[2 * e + 1];
  const float4 a = *(const float4*)(y + p0 * 128 + c);
  const float4 b = *(const float4*)(y + p1 * 128 + c);
  float4 o; o.x = a.x + b.x; o.y = a.y + b.y; o.z = a.z + b.z; o.w = a.w + b.w;
  *(float4*)(h + (long)e * 128 + c) = o;
}

// wt[n][k] = (f16) w[k][n]
__global__ void cvt_transpose_kernel(const float* __restrict__ w, _Float16* __restrict__ wt,
                                     int K, int N) {
  int i = blockIdx.x * blockDim.x + threadIdx.x;
  if (i >= K * N) return;
  int k = i / N, n = i - k * N;
  wt[n * K + k] = (_Float16)w[(long)k * N + n];
}

// Fold BN (eval) + b1 into scale/shift:  bn(t+b1) = t*s + ((b1-m)*s + bb),  s = g*rsqrt(v+eps)
__global__ void prep_bn_kernel(const float* __restrict__ b1, const float* __restrict__ g,
                               const float* __restrict__ bb, const float* __restrict__ m,
                               const float* __restrict__ v, float* __restrict__ sc,
                               float* __restrict__ sh) {
  int i = threadIdx.x;
  float s = g[i] * rsqrtf(v[i] + 1e-5f);
  sc[i] = s;
  sh[i] = (b1[i] - m[i]) * s + bb[i];
}

// ---------------------------------------------------------------------------
extern "C" void kernel_launch(void* const* d_in, const int* in_sizes, int n_in,
                              void* d_out, int out_size, void* d_ws, size_t ws_size,
                              hipStream_t stream)
{
  (void)in_sizes; (void)n_in; (void)out_size; (void)ws_size;
  const int*   ei_line = (const int*)d_in[0];    // [2, EL]
  const float* x0      = (const float*)d_in[1];  // [NN, 128]
  const int*   ei_org  = (const int*)d_in[2];    // [2, EO]
  const int*   pairs   = (const int*)d_in[3];    // [EO, 2]
  const float* iW1 = (const float*)d_in[4];
  const float* ib1 = (const float*)d_in[5];
  const float* ig  = (const float*)d_in[6];
  const float* ibb = (const float*)d_in[7];
  const float* im  = (const float*)d_in[8];
  const float* iv  = (const float*)d_in[9];
  const float* iW2 = (const float*)d_in[10];
  const float* ib2 = (const float*)d_in[11];
  const float* gW1 = (const float*)d_in[12];
  const float* gb1 = (const float*)d_in[13];
  const float* gg  = (const float*)d_in[14];
  const float* gbb = (const float*)d_in[15];
  const float* gm  = (const float*)d_in[16];
  const float* gv  = (const float*)d_in[17];
  const float* gW2 = (const float*)d_in[18];
  const float* gb2 = (const float*)d_in[19];
  const float* mW1 = (const float*)d_in[20];
  const float* mb1 = (const float*)d_in[21];
  const float* mg  = (const float*)d_in[22];
  const float* mbb = (const float*)d_in[23];
  const float* mm  = (const float*)d_in[24];
  const float* mv  = (const float*)d_in[25];
  const float* mW2 = (const float*)d_in[26];
  const float* mb2 = (const float*)d_in[27];
  const float* oW  = (const float*)d_in[28];
  const float* ob  = (const float*)d_in[29];

  // Workspace layout
  float* E0 = (float*)d_ws;                         // [EO,128] edge features
  float* E1 = E0 + (size_t)EO * 128;                // [EO,128] sum scratch (+ node bufs early)
  _Float16* w16 = (_Float16*)(E1 + (size_t)EO * 128);
  _Float16* S[11];
  for (int i = 0; i < 11; ++i) S[i] = w16 + (size_t)i * 16384; // S[10] uses 8192
  float* bn = (float*)(w16 + (size_t)10 * 16384 + 8192);       // 5 sets x {scale[128],shift[128]}

  // Node-stage buffers live inside E1 (dead before edge stage starts)
  float* xN = E1;
  float* sN = E1 + (size_t)NN * 128;
  float* yN = E1 + 2 * (size_t)NN * 128;

  const int T = 256;
  const long n4node = (long)NN * 32, n4edge = (long)EO * 32;
  const int bCopyN = (int)((n4node + T - 1) / T);
  const int bCopyE = (int)((n4edge + T - 1) / T);
  const int bScatN = (int)(((long)EO * 32 + T - 1) / T);
  const int bScatE = (int)(((long)EL * 32 + T - 1) / T);
  const int bFusedN = (NN + 127) / 128;
  const int bFusedE = (EO + 127) / 128;

  // --- weight prep: transpose + f16, fold BN ---
  cvt_transpose_kernel<<<64, T, 0, stream>>>(iW1,          S[0], 128, 128);
  cvt_transpose_kernel<<<64, T, 0, stream>>>(iW2,          S[1], 128, 128);
  cvt_transpose_kernel<<<64, T, 0, stream>>>(iW1 + 16384,  S[2], 128, 128);
  cvt_transpose_kernel<<<64, T, 0, stream>>>(iW2 + 16384,  S[3], 128, 128);
  cvt_transpose_kernel<<<64, T, 0, stream>>>(mW1,          S[4], 128, 128);
  cvt_transpose_kernel<<<64, T, 0, stream>>>(mW2,          S[5], 128, 128);
  cvt_transpose_kernel<<<64, T, 0, stream>>>(gW1,          S[6], 128, 128);
  cvt_transpose_kernel<<<64, T, 0, stream>>>(gW2,          S[7], 128, 128);
  cvt_transpose_kernel<<<64, T, 0, stream>>>(gW1 + 16384,  S[8], 128, 128);
  cvt_transpose_kernel<<<64, T, 0, stream>>>(gW2 + 16384,  S[9], 128, 128);
  cvt_transpose_kernel<<<32, T, 0, stream>>>(oW,           S[10], 128, 64);
  prep_bn_kernel<<<1, 128, 0, stream>>>(ib1,       ig,       ibb,       im,       iv,       bn + 0,    bn + 128);
  prep_bn_kernel<<<1, 128, 0, stream>>>(ib1 + 128, ig + 128, ibb + 128, im + 128, iv + 128, bn + 256,  bn + 384);
  prep_bn_kernel<<<1, 128, 0, stream>>>(mb1,       mg,       mbb,       mm,       mv,       bn + 512,  bn + 640);
  prep_bn_kernel<<<1, 128, 0, stream>>>(gb1,       gg,       gbb,       gm,       gv,       bn + 768,  bn + 896);
  prep_bn_kernel<<<1, 128, 0, stream>>>(gb1 + 128, gg + 128, gbb + 128, gm + 128, gv + 128, bn + 1024, bn + 1152);

  // --- initial GIN layer 0 (original graph) ---
  copy_vec4_kernel<<<bCopyN, T, 0, stream>>>((float4*)sN, (const float4*)x0, n4node);
  scatter_add_kernel<<<bScatN, T, 0, stream>>>(sN, x0, ei_org, ei_org + EO, EO);
  fused_mlp_kernel<<<bFusedN, T, 0, stream>>>(sN, xN, S[0], S[1], bn + 0, bn + 128, ib2, NN, 1);
  // --- initial GIN layer 1 ---
  copy_vec4_kernel<<<bCopyN, T, 0, stream>>>((float4*)sN, (const float4*)xN, n4node);
  scatter_add_kernel<<<bScatN, T, 0, stream>>>(sN, xN, ei_org, ei_org + EO, EO);
  fused_mlp_kernel<<<bFusedN, T, 0, stream>>>(sN, xN, S[2], S[3], bn + 256, bn + 384, ib2 + 128, NN, 1);
  // --- per-node MLP (rewrite: MLP once per node, then gather per edge) ---
  fused_mlp_kernel<<<bFusedN, T, 0, stream>>>(xN, yN, S[4], S[5], bn + 512, bn + 640, mb2, NN, 0);
  pair_gather_kernel<<<bScatN, T, 0, stream>>>(yN, pairs, E0, EO);
  // --- GIN layer 0 (line graph) ---
  copy_vec4_kernel<<<bCopyE, T, 0, stream>>>((float4*)E1, (const float4*)E0, n4edge);
  scatter_add_kernel<<<bScatE, T, 0, stream>>>(E1, E0, ei_line, ei_line + EL, EL);
  fused_mlp_kernel<<<bFusedE, T, 0, stream>>>(E1, E0, S[6], S[7], bn + 768, bn + 896, gb2, EO, 1);
  // --- GIN layer 1 (line graph) ---
  copy_vec4_kernel<<<bCopyE, T, 0, stream>>>((float4*)E1, (const float4*)E0, n4edge);
  scatter_add_kernel<<<bScatE, T, 0, stream>>>(E1, E0, ei_line, ei_line + EL, EL);
  fused_mlp_kernel<<<bFusedE, T, 0, stream>>>(E1, E0, S[8], S[9], bn + 1024, bn + 1152, gb2 + 128, EO, 1);
  // --- output projection ---
  out_gemm_kernel<<<bFusedE, T, 0, stream>>>(E0, (float*)d_out, S[10], ob, EO);
}
